// CxNNxNN_Attention_25409026523644
// MI455X (gfx1250) — matrix operand order repaired and checked by reading, not compiled, not run on hardware
//
#include <hip/hip_runtime.h>
#include <hip/hip_bf16.h>
#include <stdint.h>

typedef __attribute__((ext_vector_type(16))) _Float16     v16h;
typedef __attribute__((ext_vector_type(8)))  float        v8f;
typedef __attribute__((ext_vector_type(8)))  unsigned int v8u;
typedef __attribute__((ext_vector_type(4)))  unsigned int u32x4;
typedef __attribute__((ext_vector_type(8)))  unsigned int u32x8;

#define PIX 65536   // H*W = 256*256
#define HW  256

// ---------------- WMMA fragment helpers (CDNA5 16x16x32 f16 layouts) -------

// A-matrix 16x32 f16: lane L holds row M=L%16; VGPR r holds K pair
//   K = kBase + 2*(r&3) + 16*(r>>2) + 8*(L>=16)
__device__ __forceinline__ v16h frag_a(const _Float16* lds, int stride,
                                       int rowBase, int kBase) {
  int lane = threadIdx.x & 31;
  const _Float16* rp = lds + (rowBase + (lane & 15)) * stride;
  int hi8 = (lane >> 4) << 3;
  union { v8u u; v16h h; } c;
#pragma unroll
  for (int r = 0; r < 8; ++r) {
    int k = kBase + ((r & 3) << 1) + ((r >> 2) << 4) + hi8;
    c.u[r] = *(const unsigned int*)(rp + k);   // dword = 2 contiguous halves
  }
  return c.h;
}

// B-matrix 32x16 f16 (LDS stored N-major: [N][K]): lane L holds col N=L%16;
//   VGPR r holds K pair K = kBase + 2*(r&3) + 16*(r>>2) + 8*(L>=16)
__device__ __forceinline__ v16h frag_b(const _Float16* lds, int stride,
                                       int rowBase, int kBase) {
  int lane = threadIdx.x & 31;
  const _Float16* rp = lds + (rowBase + (lane & 15)) * stride;
  int hi8 = (lane >> 4) << 3;
  union { v8u u; v16h h; } c;
#pragma unroll
  for (int r = 0; r < 8; ++r) {
    int k = kBase + ((r & 3) << 1) + ((r >> 2) << 4) + hi8;
    c.u[r] = *(const unsigned int*)(rp + k);
  }
  return c.h;
}

__device__ __forceinline__ v8f wmma_f16(v16h a, v16h b, v8f c) {
  return __builtin_amdgcn_wmma_f32_16x16x32_f16(false, a, false, b,
                                                (short)0, c, false, false);
}

__device__ __forceinline__ v8f zero8() {
  v8f z = {0.f, 0.f, 0.f, 0.f, 0.f, 0.f, 0.f, 0.f};
  return z;
}

// token t=(nh,nw), feature d=(hh,ww) -> pixel (hh*8+nh)*256 + ww*8+nw
__device__ __forceinline__ int pix(int t, int d) {
  return ((d >> 5) << 11) + ((t >> 3) << 8) + ((d & 31) << 3) + (t & 7);
}

// ---------------- Tensor Data Mover: 1-D f16 slab -> LDS -------------------
// D# per CDNA5 ISA. group0: count=1, lds_addr,
// 57b global addr, type=2. group1: data_size=2B, tensor_dim0=tile_dim0=nelem,
// optional LDS padding of 1 dword per 32 dwords (row stride 66 halves).
__device__ __forceinline__ unsigned int lds_off(const void* p) {
  return (unsigned int)(unsigned long long)(uintptr_t)p;   // low 32b = LDS offset
}

__device__ __forceinline__ void tdm_load_1d(unsigned int ldsOff,
                                            const void* gsrc,
                                            unsigned int nelem,   // f16 elems
                                            int padRows) {
  unsigned long long ga = (unsigned long long)(uintptr_t)gsrc;
  u32x4 g0;
  g0[0] = 1u;                                               // count=1
  g0[1] = ldsOff;
  g0[2] = (unsigned int)ga;
  g0[3] = ((unsigned int)(ga >> 32) & 0x01FFFFFFu) | 0x80000000u;  // type=2
  unsigned int d0 = 0x00010000u;                            // data_size=2B
  if (padRows) d0 |= (1u << 20) | (4u << 22);               // pad 1dw / 32dw
  u32x8 g1;
  g1[0] = d0;
  g1[1] = (nelem & 0xFFFFu) << 16;                          // tensor_dim0 lo
  g1[2] = (nelem >> 16) | (1u << 16);                       // dim0 hi, dim1=1
  g1[3] = (nelem & 0xFFFFu) << 16;                          // tile_dim0
  g1[4] = 0u;                                               // tile_dim1/2 unused
  g1[5] = nelem;                                            // dim0 stride
  g1[6] = 0u;
  g1[7] = 0u;
  asm volatile("tensor_load_to_lds %0, %1" :: "s"(g0), "s"(g1) : "memory");
}

__device__ __forceinline__ void tdm_wait0() {
  __builtin_amdgcn_s_wait_tensorcnt(0);
}

// ---------------- Kernel 1: qkv = W_qkv(192x64) @ x, f32 in -> f16 out -----
__global__ void __launch_bounds__(256) k_qkv(const float* __restrict__ x,
                                             const float* __restrict__ wq,
                                             _Float16* __restrict__ qkv) {
  __shared__ _Float16 Wl[192 * 66];
  __shared__ _Float16 Xl[64 * 66];
  int tid = threadIdx.x;
  int b = blockIdx.x >> 10;
  int pbase = (blockIdx.x & 1023) << 6;
#pragma unroll
  for (int i = 0; i < 48; ++i) {
    int idx = tid + (i << 8);                 // 192*64 weights
    Wl[(idx >> 6) * 66 + (idx & 63)] = (_Float16)wq[idx];
  }
  const float* xb = x + (size_t)b * 64 * PIX + pbase;
#pragma unroll
  for (int i = 0; i < 16; ++i) {
    int idx = tid + (i << 8);                 // 64 chan x 64 pixels
    int c = idx >> 6, pl = idx & 63;
    Xl[pl * 66 + c] = (_Float16)xb[(size_t)c * PIX + pl];   // N-major [p][c]
  }
  __syncthreads();
  int wave = tid >> 5, lane = tid & 31;
  for (int tile = wave; tile < 48; tile += 8) {    // 12 Mtiles x 4 Ntiles
    int Mt = tile >> 2, Nt = tile & 3;
    v8f acc = zero8();
#pragma unroll
    for (int kb = 0; kb < 64; kb += 32)
      acc = wmma_f16(frag_a(Wl, 66, Mt * 16, kb),
                     frag_b(Xl, 66, Nt * 16, kb), acc);
    int p = pbase + Nt * 16 + (lane & 15);
    int obase = Mt * 16 + ((lane >> 4) << 3);
#pragma unroll
    for (int j = 0; j < 8; ++j)
      qkv[((size_t)b * 192 + obase + j) * PIX + p] = (_Float16)acc[j];
  }
}

// ---------------- Kernel 2: 3x3 depthwise, zero pad, f16 -> f16 ------------
__global__ void __launch_bounds__(256) k_dw(const _Float16* __restrict__ in,
                                            const float* __restrict__ wdw,
                                            _Float16* __restrict__ out) {
  int bo = blockIdx.y;                 // b*192 + channel
  int ch = bo % 192;
  int p = (blockIdx.x << 8) + threadIdx.x;
  int h = p >> 8, w = p & 255;
  const float* wc = wdw + ch * 9;
  const _Float16* src = in + (size_t)bo * PIX;
  __builtin_prefetch(src + (((h + 4) & 255) * HW + w), 0, 1);  // global_prefetch_b8
  float s = 0.f;
#pragma unroll
  for (int dy = -1; dy <= 1; ++dy) {
    int hh = h + dy;
    if (hh < 0 || hh > 255) continue;
#pragma unroll
    for (int dx = -1; dx <= 1; ++dx) {
      int ww = w + dx;
      if (ww < 0 || ww > 255) continue;
      s += wc[(dy + 1) * 3 + dx + 1] * (float)src[hh * HW + ww];
    }
  }
  out[(size_t)bo * PIX + p] = (_Float16)s;
}

// ---------------- Kernel 3: per-slice attention (one block per slice) ------
// Q,K,V: 64 tokens x 1024 features.  Each 64-feature chunk is a contiguous
// 8KB slab (16 image rows) -> staged by TDM, rearranged in LDS.  Row norms
// folded into logit scaling:  softmax(diag(sq) QK^T diag(sk) * temp).
__global__ void __launch_bounds__(256) k_attn(const _Float16* __restrict__ qkv,
                                              const float* __restrict__ temperature,
                                              _Float16* __restrict__ aout) {
  __shared__ _Float16 Ql[64 * 66];     // attn phase: Q [t][d]; out phase: V [d][t]
  __shared__ _Float16 Kl[64 * 66];
  __shared__ _Float16 Al[64 * 66];     // softmaxed attn f16 [t][k]
  __shared__ float    At[64 * 66];     // logits f32
  __shared__ _Float16 rawQ[4096];      // TDM slab (also rawV in out phase)
  __shared__ _Float16 rawK[4096];
  __shared__ float    red[512];
  __shared__ float    scl[128];        // [0:64) 1/||q||, [64:128) 1/||k||
  int tid = threadIdx.x;
  int wave = tid >> 5, lane = tid & 31;
  int b = blockIdx.x >> 6;
  int ch = blockIdx.x & 63;
  float temp = temperature[ch >> 3];
  const _Float16* Q = qkv + ((size_t)b * 192 + ch) * PIX;
  const _Float16* K = qkv + ((size_t)b * 192 + 64 + ch) * PIX;
  const _Float16* V = qkv + ((size_t)b * 192 + 128 + ch) * PIX;
  _Float16* O = aout + (size_t)b * PIX * 64 + ch;   // pixel-major [p][64]

  int Mt0 = (wave >> 2) * 16, Mt1 = Mt0 + 32, Nt = (wave & 3) * 16;

  // ---- attn = QK^T over 16 chunks of 64 features, TDM double-issue ----
  if (wave == 0) {
    tdm_load_1d(lds_off(rawQ), Q, 4096, 0);
    tdm_load_1d(lds_off(rawK), K, 4096, 0);
  }
  float sq = 0.f, sk = 0.f;
  v8f acc0 = zero8(), acc1 = zero8();
  for (int chunk = 0; chunk < 16; ++chunk) {
    if (wave == 0) tdm_wait0();
    __syncthreads();                       // slabs ready; Ql/Kl consumers done
#pragma unroll
    for (int i = 0; i < 16; ++i) {         // slab [16 rows x 256] -> [t][d]
      int e = tid + (i << 8);
      int t = e & 63, dl = e >> 6;
      int r = ((((dl >> 5) << 3) + (t >> 3)) << 8) + ((dl & 31) << 3) + (t & 7);
      _Float16 qh = rawQ[r], kh = rawK[r];
      float qf = (float)qh, kf = (float)kh;
      sq += qf * qf; sk += kf * kf;        // thread owns token t = tid&63
      Ql[t * 66 + dl] = qh;
      Kl[t * 66 + dl] = kh;
    }
    __syncthreads();                       // raw free; Ql/Kl published
    if (wave == 0 && chunk < 15) {         // DMA next slabs under the WMMAs
      tdm_load_1d(lds_off(rawQ), Q + (chunk + 1) * 4096, 4096, 0);
      tdm_load_1d(lds_off(rawK), K + (chunk + 1) * 4096, 4096, 0);
    }
#pragma unroll
    for (int kb = 0; kb < 64; kb += 32) {
      v16h bf = frag_b(Kl, 66, Nt, kb);
      acc0 = wmma_f16(frag_a(Ql, 66, Mt0, kb), bf, acc0);
      acc1 = wmma_f16(frag_a(Ql, 66, Mt1, kb), bf, acc1);
    }
  }

  // ---- row norms from folded sums; logits with scaling ----
  red[tid] = sq; red[256 + tid] = sk;
  __syncthreads();
  if (tid < 64) {
    float s = red[tid] + red[tid + 64] + red[tid + 128] + red[tid + 192];
    scl[tid] = 1.f / fmaxf(sqrtf(s), 1e-12f);
    float s2 = red[256 + tid] + red[320 + tid] + red[384 + tid] + red[448 + tid];
    scl[64 + tid] = 1.f / fmaxf(sqrtf(s2), 1e-12f);
  }
  __syncthreads();
  {
    int col = Nt + (lane & 15);
    int r0 = Mt0 + ((lane >> 4) << 3);
    float ksc = scl[64 + col] * temp;
#pragma unroll
    for (int j = 0; j < 8; ++j) {
      At[(r0 + j) * 66 + col]      = acc0[j] * scl[r0 + j] * ksc;
      At[(r0 + 32 + j) * 66 + col] = acc1[j] * scl[r0 + 32 + j] * ksc;
    }
  }
  __syncthreads();

  // ---- softmax rows (one thread per row) ----
  if (tid < 64) {
    float m = -3.0e38f;
    for (int n = 0; n < 64; ++n) m = fmaxf(m, At[tid * 66 + n]);
    float s = 0.f;
    for (int n = 0; n < 64; ++n) {
      float e = __expf(At[tid * 66 + n] - m);
      At[tid * 66 + n] = e; s += e;
    }
    float inv = 1.f / s;
    for (int n = 0; n < 64; ++n)
      Al[tid * 66 + n] = (_Float16)(At[tid * 66 + n] * inv);
  }
  __syncthreads();

  // ---- out = attn @ V, feature dim in 16 chunks, TDM-staged V ----
  _Float16* Vl = Ql;                       // reuse (Q dead)
  _Float16* rawV = rawQ;
  v16h oa00 = frag_a(Al, 66, Mt0, 0);
  v16h oa01 = frag_a(Al, 66, Mt0, 32);
  v16h oa10 = frag_a(Al, 66, Mt1, 0);
  v16h oa11 = frag_a(Al, 66, Mt1, 32);
  if (wave == 0) tdm_load_1d(lds_off(rawV), V, 4096, 0);
  for (int chunk = 0; chunk < 16; ++chunk) {
    if (wave == 0) tdm_wait0();
    __syncthreads();
#pragma unroll
    for (int i = 0; i < 16; ++i) {         // slab -> V [d][t] (feature-major)
      int e = tid + (i << 8);
      int t = e & 63, dl = e >> 6;
      int r = ((((dl >> 5) << 3) + (t >> 3)) << 8) + ((dl & 31) << 3) + (t & 7);
      Vl[dl * 66 + t] = rawV[r];
    }
    __syncthreads();
    if (wave == 0 && chunk < 15)
      tdm_load_1d(lds_off(rawV), V + (chunk + 1) * 4096, 4096, 0);
    v16h vb0 = frag_b(Vl, 66, Nt, 0);
    v16h vb1 = frag_b(Vl, 66, Nt, 32);
    v8f o0 = zero8(), o1 = zero8();
    o0 = wmma_f16(oa00, vb0, o0);
    o0 = wmma_f16(oa01, vb1, o0);
    o1 = wmma_f16(oa10, vb0, o1);
    o1 = wmma_f16(oa10 == oa10 ? oa11 : oa11, vb1, o1);   // (keep simple)
    int dcol = (chunk << 6) + Nt + (lane & 15);
    int t0 = Mt0 + ((lane >> 4) << 3);
#pragma unroll
    for (int j = 0; j < 8; ++j) {
      O[(size_t)pix(t0 + j, dcol) * 64]      = (_Float16)o0[j];
      O[(size_t)pix(t0 + 32 + j, dcol) * 64] = (_Float16)o1[j];
    }
  }
}

// ---------------- Kernel 4: out = W_proj(64x64) @ feat, f16 in -> f32 out --
// Input is pixel-major [B][PIX][64]: the 64px x 64ch tile is one contiguous
// 8KB slab -> TDM load with LDS padding gives stride-66 [p][c] directly.
__global__ void __launch_bounds__(256) k_proj(const _Float16* __restrict__ fin,
                                              const float* __restrict__ wp,
                                              float* __restrict__ out) {
  __shared__ _Float16 Wl[64 * 66];
  __shared__ _Float16 Xl[64 * 66];
  int tid = threadIdx.x;
  int b = blockIdx.x >> 10;
  int pbase = (blockIdx.x & 1023) << 6;
  if ((tid >> 5) == 0)
    tdm_load_1d(lds_off(Xl), fin + ((size_t)b * PIX + pbase) * 64, 4096, 1);
#pragma unroll
  for (int i = 0; i < 16; ++i) {
    int idx = tid + (i << 8);
    Wl[(idx >> 6) * 66 + (idx & 63)] = (_Float16)wp[idx];
  }
  if ((tid >> 5) == 0) tdm_wait0();
  __syncthreads();
  int wave = tid >> 5, lane = tid & 31;
  for (int tile = wave; tile < 16; tile += 8) {
    int Mt = tile >> 2, Nt = tile & 3;
    v8f acc = zero8();
#pragma unroll
    for (int kb = 0; kb < 64; kb += 32)
      acc = wmma_f16(frag_a(Wl, 66, Mt * 16, kb),
                     frag_b(Xl, 66, Nt * 16, kb), acc);
    int p = pbase + Nt * 16 + (lane & 15);
    int obase = Mt * 16 + ((lane >> 4) << 3);
#pragma unroll
    for (int j = 0; j < 8; ++j)
      out[((size_t)b * 64 + obase + j) * PIX + p] = acc[j];
  }
}

// ---------------------------------------------------------------------------
extern "C" void kernel_launch(void* const* d_in, const int* in_sizes, int n_in,
                              void* d_out, int out_size, void* d_ws, size_t ws_size,
                              hipStream_t stream) {
  (void)in_sizes; (void)n_in; (void)out_size; (void)ws_size;
  const float* x    = (const float*)d_in[0];   // [8,64,256,256]
  const float* wqkv = (const float*)d_in[1];   // [192,64]
  const float* wdw  = (const float*)d_in[2];   // [192,1,3,3]
  const float* wprj = (const float*)d_in[3];   // [64,64]
  const float* temp = (const float*)d_in[4];   // [8]
  float* out = (float*)d_out;                  // [8,64,256,256] f32

  // workspace: two f16 buffers of [8,192,65536]; attention output (pixel-
  // major [8][65536][64]) reuses buffer A, which is dead after k_dw.
  _Float16* qkvA = (_Float16*)d_ws;                       // 201,326,592 B
  _Float16* qkvB = qkvA + (size_t)8 * 192 * PIX;          // 201,326,592 B
  _Float16* aout = qkvA;

  k_qkv <<<dim3(8192),      dim3(256), 0, stream>>>(x, wqkv, qkvA);
  k_dw  <<<dim3(256, 1536), dim3(256), 0, stream>>>(qkvA, wdw, qkvB);
  k_attn<<<dim3(512),       dim3(256), 0, stream>>>(qkvB, temp, aout);
  k_proj<<<dim3(8192),      dim3(256), 0, stream>>>(aout, wprj, out);
}
